// PrototypeNet_48773648613464
// MI455X (gfx1250) — matrix operand order repaired and checked by reading, not compiled, not run on hardware
//
#include <hip/hip_runtime.h>
#include <hip/hip_bf16.h>
#include <math.h>

typedef __bf16 bf16;
typedef __attribute__((ext_vector_type(8)))  __bf16 bf16x8;
typedef __attribute__((ext_vector_type(16))) __bf16 bf16x16;
typedef __attribute__((ext_vector_type(8)))  float  f32x8;

constexpr int   Bv   = 8192;     // batch per view
constexpr int   TOT  = 16384;    // both views concatenated
constexpr int   NZ   = 1024;
constexpr int   Hd   = 4096;
constexpr int   Kp   = 4096;     // prototypes (== Hd here)
constexpr float TEMP_INV = 5.0f; // 1 / 0.2
constexpr float BETA = 0.25f;

// ---------------------------------------------------------------------------
// bf16 WMMA GEMM:  C[M,N] = act(A[M,Kd] @ Bm[N,Kd]^T + bias)
// 128x128 tile per 256-thread block (8 waves, each 32x64).
// Global->LDS uses CDNA5 async-LDS copies (ASYNCcnt) with ping-pong buffers,
// so tile k+1 streams in while WMMAs consume tile k.
// ---------------------------------------------------------------------------
template <bool RELU, bool OUT_BF16>
__global__ __launch_bounds__(256) void gemm_bf16_wmma(
    const bf16* __restrict__ A, const bf16* __restrict__ Bm,
    const float* __restrict__ bias, float* __restrict__ Cf,
    bf16* __restrict__ Cb, int N, int Kd)
{
    __shared__ bf16 As[2][128 * 32];
    __shared__ bf16 Bs[2][128 * 32];

    const int tid  = threadIdx.x;
    const int lane = tid & 31;
    const int wv   = tid >> 5;
    const int wm   = (wv & 3) * 32;   // wave M offset in tile
    const int wn   = (wv >> 2) * 64;  // wave N offset in tile
    const int half = lane >> 4;       // 0 or 1
    const int l16  = lane & 15;

    const int bm = blockIdx.x * 128;
    const int bn = blockIdx.y * 128;

    f32x8 acc[2][4];
#pragma unroll
    for (int mi = 0; mi < 2; ++mi)
#pragma unroll
        for (int ni = 0; ni < 4; ++ni)
#pragma unroll
            for (int r = 0; r < 8; ++r) acc[mi][ni][r] = 0.0f;

    const int ktiles = Kd >> 5;

    // Issue the async global->LDS copies for one 128x32 A tile + 128x32 B tile.
    // Each thread moves 2x16B of A and 2x16B of B; no VGPR staging.
    auto issue_tile = [&](int kt, int buf) {
#pragma unroll
        for (int i = 0; i < 2; ++i) {
            const int idx = tid + i * 256;      // 0..511
            const int r   = idx >> 2;           // 0..127
            const int c8  = (idx & 3) * 8;      // 0,8,16,24 (bf16 elems)
            const bf16* ga = A  + (size_t)(bm + r) * Kd + kt * 32 + c8;
            const bf16* gb = Bm + (size_t)(bn + r) * Kd + kt * 32 + c8;
            const uint32_t la = (uint32_t)(size_t)(const void*)&As[buf][r * 32 + c8];
            const uint32_t lb = (uint32_t)(size_t)(const void*)&Bs[buf][r * 32 + c8];
            asm volatile("global_load_async_to_lds_b128 %0, %1, off"
                         :: "v"(la), "v"(ga) : "memory");
            asm volatile("global_load_async_to_lds_b128 %0, %1, off"
                         :: "v"(lb), "v"(gb) : "memory");
        }
    };

    issue_tile(0, 0);
    for (int kt = 0; kt < ktiles; ++kt) {
        const int buf = kt & 1;
        // Tile kt fully landed in LDS (ASYNCcnt) and visible to all waves.
        asm volatile("s_wait_asynccnt 0" ::: "memory");
        __syncthreads();
        // Stream tile kt+1 into the other buffer while we compute tile kt.
        if (kt + 1 < ktiles) issue_tile(kt + 1, buf ^ 1);

        // ---- fragments per ISA layout ----
        bf16x16 afrag[2], bfrag[4];
#pragma unroll
        for (int mi = 0; mi < 2; ++mi) {
            const bf16* p = &As[buf][(wm + mi * 16 + l16) * 32 + half * 8];
            bf16x8 lo = *(const bf16x8*)p;
            bf16x8 hi = *(const bf16x8*)(p + 16);
            afrag[mi] = __builtin_shufflevector(lo, hi, 0, 1, 2, 3, 4, 5, 6, 7,
                                                8, 9, 10, 11, 12, 13, 14, 15);
        }
#pragma unroll
        for (int ni = 0; ni < 4; ++ni) {
            const bf16* p = &Bs[buf][(wn + ni * 16 + l16) * 32 + half * 8];
            bf16x8 lo = *(const bf16x8*)p;
            bf16x8 hi = *(const bf16x8*)(p + 16);
            bfrag[ni] = __builtin_shufflevector(lo, hi, 0, 1, 2, 3, 4, 5, 6, 7,
                                                8, 9, 10, 11, 12, 13, 14, 15);
        }
#pragma unroll
        for (int mi = 0; mi < 2; ++mi)
#pragma unroll
            for (int ni = 0; ni < 4; ++ni)
                acc[mi][ni] = __builtin_amdgcn_wmma_f32_16x16x32_bf16(
                    false, afrag[mi], false, bfrag[ni], (short)0, acc[mi][ni],
                    false, false);
        // No trailing barrier: each wave's LDS reads completed before its
        // next-iteration top barrier, which gates the buffer's async refill.
    }

    // ---- epilogue ----
#pragma unroll
    for (int mi = 0; mi < 2; ++mi) {
#pragma unroll
        for (int ni = 0; ni < 4; ++ni) {
            const int col = bn + wn + ni * 16 + l16;
            const float bvv = bias ? bias[col] : 0.0f;
#pragma unroll
            for (int r = 0; r < 8; ++r) {
                const int row = bm + wm + mi * 16 + half * 8 + r;
                float x = acc[mi][ni][r] + bvv;
                if (RELU) x = fmaxf(x, 0.0f);
                if constexpr (OUT_BF16)
                    Cb[(size_t)row * N + col] = (bf16)x;
                else
                    Cf[(size_t)row * N + col] = x;
            }
        }
    }
}

// ---------------------------------------------------------------------------
// Elementwise / reduction helper kernels
// ---------------------------------------------------------------------------
__global__ __launch_bounds__(256) void f32_to_bf16_kernel(
    const float* __restrict__ in, bf16* __restrict__ out, long n)
{
    long i = (long)blockIdx.x * blockDim.x + threadIdx.x;
    const long st = (long)gridDim.x * blockDim.x;
    for (; i < n; i += st) out[i] = (bf16)in[i];
}

__global__ __launch_bounds__(256) void exp_inplace_kernel(float* p, long n)
{
    long i = (long)blockIdx.x * blockDim.x + threadIdx.x;
    const long st = (long)gridDim.x * blockDim.x;
    for (; i < n; i += st) p[i] = expf(p[i] * 20.0f);  // exp(P / 0.05)
}

__global__ void fill_kernel(float* p, int n, float val)
{
    int i = blockIdx.x * blockDim.x + threadIdx.x;
    if (i < n) p[i] = val;
}

// Row L2-normalize: outf = in/max(||row||,1e-12); optional bf16 copy.
// in/outf may alias (each element read+written by the same thread only).
__global__ __launch_bounds__(256) void rownorm_kernel(
    const float* in, float* outf, bf16* outb, int D)
{
    __shared__ float sm[256];
    const int t = threadIdx.x;
    const size_t base = (size_t)blockIdx.x * D;
    float acc = 0.0f;
    for (int i = t; i < D; i += 256) { float x = in[base + i]; acc += x * x; }
    sm[t] = acc; __syncthreads();
    for (int s = 128; s > 0; s >>= 1) { if (t < s) sm[t] += sm[t + s]; __syncthreads(); }
    const float inv = 1.0f / fmaxf(sqrtf(sm[0]), 1e-12f);
    __syncthreads();
    for (int i = t; i < D; i += 256) {
        const float x = in[base + i] * inv;
        outf[base + i] = x;
        if (outb) outb[base + i] = (bf16)x;
    }
}

// cs[view*Kp+k] += sum_{r in chunk} E[view][r][k] * v[view][r]
__global__ __launch_bounds__(256) void colsum_kernel(
    const float* __restrict__ E, const float* __restrict__ v, float* cs)
{
    const int gk = blockIdx.x * 256 + threadIdx.x;  // 0..2*Kp-1
    const int view = gk >> 12;
    const int k = gk & (Kp - 1);
    const int r0 = blockIdx.y * 512;
    const float* vv = v + view * Bv;
    size_t base = (size_t)(view * Bv + r0) * Kp + k;
    float acc = 0.0f;
    for (int r = 0; r < 512; ++r) acc += E[base + (size_t)r * Kp] * vv[r0 + r];
    atomicAdd(&cs[gk], acc);
}

__global__ __launch_bounds__(256) void ukern_kernel(const float* cs, float* u)
{
    const int i = blockIdx.x * 256 + threadIdx.x;
    u[i] = (1.0f / (float)Kp) / cs[i];
}

// v[g] = (1/Bv) / sum_k E[g][k]*u[view][k]
__global__ __launch_bounds__(256) void rowsumv_kernel(
    const float* __restrict__ E, const float* __restrict__ u, float* v)
{
    __shared__ float sm[256];
    const int g = blockIdx.x, t = threadIdx.x;
    const int view = g >> 13;
    const size_t rb = (size_t)g * Kp;
    const float* ur = u + view * Kp;
    float acc = 0.0f;
    for (int k = t; k < Kp; k += 256) acc += E[rb + k] * ur[k];
    sm[t] = acc; __syncthreads();
    for (int s = 128; s > 0; s >>= 1) { if (t < s) sm[t] += sm[t + s]; __syncthreads(); }
    if (t == 0) v[g] = (1.0f / (float)Bv) / sm[0];
}

// Per-row max/argmax/sum-exp of q = Bv*u*E*v (both views), for softmax+VQ.
__global__ __launch_bounds__(256) void rowstats_kernel(
    const float* __restrict__ E, const float* __restrict__ u,
    const float* __restrict__ v, float* qmax, float* sexp, int* amax)
{
    __shared__ float sv[256];
    __shared__ int si[256];
    const int b = blockIdx.x, t = threadIdx.x;
    for (int view = 0; view < 2; ++view) {
        const size_t rb = (size_t)(view * Bv + b) * Kp;
        const float scale = (float)Bv * v[view * Bv + b];
        const float* ur = u + view * Kp;
        float mval = -1.0f; int midx = 0;
        for (int k = t; k < Kp; k += 256) {
            const float q = scale * ur[k] * E[rb + k];
            if (q > mval) { mval = q; midx = k; }
        }
        sv[t] = mval; si[t] = midx; __syncthreads();
        for (int s = 128; s > 0; s >>= 1) {
            if (t < s) {
                if (sv[t + s] > sv[t] || (sv[t + s] == sv[t] && si[t + s] < si[t])) {
                    sv[t] = sv[t + s]; si[t] = si[t + s];
                }
            }
            __syncthreads();
        }
        const float qm = sv[0]; const int qi = si[0];
        __syncthreads();
        float acc = 0.0f;
        for (int k = t; k < Kp; k += 256) {
            const float q = scale * ur[k] * E[rb + k];
            acc += expf((q - qm) * TEMP_INV);
        }
        sv[t] = acc; __syncthreads();
        for (int s = 128; s > 0; s >>= 1) { if (t < s) sv[t] += sv[t + s]; __syncthreads(); }
        if (t == 0) {
            qmax[view * Bv + b] = qm;
            sexp[view * Bv + b] = sv[0];
            amax[view * Bv + b] = qi;
        }
        __syncthreads();
    }
}

// accbuf[0] += sum_k q1*(-log(p2+1e-8)) + q2*(-log(p1+1e-8)) for this row.
__global__ __launch_bounds__(256) void cross_kernel(
    const float* __restrict__ E, const float* __restrict__ u,
    const float* __restrict__ v, const float* qmax, const float* sexp,
    float* accbuf)
{
    __shared__ float sm[256];
    const int b = blockIdx.x, t = threadIdx.x;
    const size_t rb1 = (size_t)b * Kp;
    const size_t rb2 = (size_t)(Bv + b) * Kp;
    const float sc1 = (float)Bv * v[b];
    const float sc2 = (float)Bv * v[Bv + b];
    const float m1 = qmax[b] * TEMP_INV, m2 = qmax[Bv + b] * TEMP_INV;
    const float is1 = 1.0f / sexp[b], is2 = 1.0f / sexp[Bv + b];
    float acc = 0.0f;
    for (int k = t; k < Kp; k += 256) {
        const float q1 = sc1 * u[k] * E[rb1 + k];
        const float q2 = sc2 * u[Kp + k] * E[rb2 + k];
        const float p1 = expf(q1 * TEMP_INV - m1) * is1;
        const float p2 = expf(q2 * TEMP_INV - m2) * is2;
        acc += q1 * logf(p2 + 1e-8f) + q2 * logf(p1 + 1e-8f);
    }
    sm[t] = acc; __syncthreads();
    for (int s = 128; s > 0; s >>= 1) { if (t < s) sm[t] += sm[t + s]; __syncthreads(); }
    if (t == 0) atomicAdd(&accbuf[0], -sm[0]);
}

// accbuf[1] += ||wn[argmax] - zp[row]||^2
__global__ __launch_bounds__(256) void vq_kernel(
    const float* __restrict__ wn, const float* __restrict__ zp,
    const int* __restrict__ amax, float* accbuf)
{
    __shared__ float sm[256];
    const int g = blockIdx.x, t = threadIdx.x;
    const int idx = amax[g];
    const float* wrow = wn + (size_t)idx * Hd;
    const float* zrow = zp + (size_t)g * Hd;
    float acc = 0.0f;
    for (int d = t; d < Hd; d += 256) {
        const float df = wrow[d] - zrow[d];
        acc += df * df;
    }
    sm[t] = acc; __syncthreads();
    for (int s = 128; s > 0; s >>= 1) { if (t < s) sm[t] += sm[t + s]; __syncthreads(); }
    if (t == 0) atomicAdd(&accbuf[1], sm[0]);
}

__global__ void finalize_kernel(const float* accbuf, float* out)
{
    out[0] = accbuf[0] / (2.0f * (float)Bv);
    out[1] = (1.0f + BETA) * accbuf[1] / ((float)Bv * (float)Hd);
}

// ---------------------------------------------------------------------------
extern "C" void kernel_launch(void* const* d_in, const int* in_sizes, int n_in,
                              void* d_out, int out_size, void* d_ws, size_t ws_size,
                              hipStream_t stream)
{
    (void)in_sizes; (void)n_in; (void)out_size; (void)ws_size;
    const float* z1 = (const float*)d_in[0];
    const float* z2 = (const float*)d_in[1];
    const float* W1 = (const float*)d_in[2];
    const float* b1 = (const float*)d_in[3];
    const float* W2 = (const float*)d_in[4];
    const float* b2 = (const float*)d_in[5];
    const float* pw = (const float*)d_in[6];
    float* out = (float*)d_out;

    char* ws = (char*)d_ws;
    size_t off = 0;
    auto alloc = [&](size_t bytes) -> void* {
        void* p = ws + off;
        off = (off + bytes + 255) & ~(size_t)255;
        return p;
    };
    bf16*  zc   = (bf16*) alloc((size_t)TOT * NZ * 2);
    bf16*  w1b  = (bf16*) alloc((size_t)Hd * NZ * 2);
    bf16*  w2b  = (bf16*) alloc((size_t)Kp * Hd * 2);
    bf16*  hb   = (bf16*) alloc((size_t)TOT * Hd * 2);   // h, later zp (bf16)
    float* zpf  = (float*)alloc((size_t)TOT * Hd * 4);   // mlp out -> zp (f32)
    float* wnf  = (float*)alloc((size_t)Kp * Hd * 4);
    bf16*  wnb  = (bf16*) alloc((size_t)Kp * Hd * 2);
    float* Ebuf = (float*)alloc((size_t)TOT * Kp * 4);   // P -> E (in place)
    float* u    = (float*)alloc((size_t)2 * Kp * 4);
    float* v    = (float*)alloc((size_t)2 * Bv * 4);
    float* cs   = (float*)alloc((size_t)2 * Kp * 4);
    float* qmax = (float*)alloc((size_t)2 * Bv * 4);
    float* sexp = (float*)alloc((size_t)2 * Bv * 4);
    int*   amax = (int*)  alloc((size_t)2 * Bv * 4);
    float* accb = (float*)alloc(256);

    const dim3 blk(256);

    // precision conversion + prototype normalization
    f32_to_bf16_kernel<<<2048, blk, 0, stream>>>(z1, zc, (long)Bv * NZ);
    f32_to_bf16_kernel<<<2048, blk, 0, stream>>>(z2, zc + (size_t)Bv * NZ, (long)Bv * NZ);
    f32_to_bf16_kernel<<<2048, blk, 0, stream>>>(W1, w1b, (long)Hd * NZ);
    f32_to_bf16_kernel<<<2048, blk, 0, stream>>>(W2, w2b, (long)Kp * Hd);
    rownorm_kernel<<<Kp, blk, 0, stream>>>(pw, wnf, wnb, Hd);

    // GEMM1: h = relu(z @ W1^T + b1)   [bf16 out]
    gemm_bf16_wmma<true, true><<<dim3(TOT / 128, Hd / 128), blk, 0, stream>>>(
        zc, w1b, b1, nullptr, hb, Hd, NZ);
    // GEMM2: mlp = h @ W2^T + b2       [f32 out]
    gemm_bf16_wmma<false, false><<<dim3(TOT / 128, Kp / 128), blk, 0, stream>>>(
        hb, w2b, b2, zpf, nullptr, Kp, Hd);
    // zp = normalize(mlp) rows; bf16 copy overwrites hb
    rownorm_kernel<<<TOT, blk, 0, stream>>>(zpf, zpf, hb, Hd);
    // GEMM3: P = zp @ wn^T             [f32 out]
    gemm_bf16_wmma<false, false><<<dim3(TOT / 128, Kp / 128), blk, 0, stream>>>(
        hb, wnb, nullptr, Ebuf, nullptr, Kp, Hd);
    // E = exp(P / 0.05)
    exp_inplace_kernel<<<4096, blk, 0, stream>>>(Ebuf, (long)TOT * Kp);

    // Sinkhorn (scaling-vector form, exact match to reference)
    fill_kernel<<<(2 * Bv + 255) / 256, blk, 0, stream>>>(v, 2 * Bv, 1.0f);
    fill_kernel<<<1, blk, 0, stream>>>(accb, 2, 0.0f);
    for (int it = 0; it < 3; ++it) {
        fill_kernel<<<(2 * Kp + 255) / 256, blk, 0, stream>>>(cs, 2 * Kp, 0.0f);
        colsum_kernel<<<dim3(2 * Kp / 256, 16), blk, 0, stream>>>(Ebuf, v, cs);
        ukern_kernel<<<2 * Kp / 256, blk, 0, stream>>>(cs, u);
        rowsumv_kernel<<<TOT, blk, 0, stream>>>(Ebuf, u, v);
    }

    // losses
    rowstats_kernel<<<Bv, blk, 0, stream>>>(Ebuf, u, v, qmax, sexp, amax);
    cross_kernel<<<Bv, blk, 0, stream>>>(Ebuf, u, v, qmax, sexp, accb);
    vq_kernel<<<TOT, blk, 0, stream>>>(wnf, zpf, amax, accb);
    finalize_kernel<<<1, 1, 0, stream>>>(accb, out);
}
